// GATv2Classifier_69999376990329
// MI455X (gfx1250) — matrix operand order repaired and checked by reading, not compiled, not run on hardware
//
#include <hip/hip_runtime.h>

// ---------------- problem constants (match reference) ----------------
#define N_NODES 50000
#define E_EDGES 400000
#define B_GRAPHS 16
#define FEAT 256          // IN_DIM == H*D == 256 for both layers
#define DHEAD 64
#define NHEAD 4
#define NCLS 10
#define NEG_SLOPE 0.2f

typedef __attribute__((ext_vector_type(16))) __bf16 v16bf;
typedef __attribute__((ext_vector_type(8)))  float  v8f;

// ---------------- helpers ----------------
__device__ __forceinline__ unsigned short f32_to_bf16(float f) {
    unsigned int u = __float_as_uint(f);
    u += 0x7FFFu + ((u >> 16) & 1u);      // round-to-nearest-even
    return (unsigned short)(u >> 16);
}

// float atomic max via integer compare trick (works for any sign, init = -inf)
__device__ __forceinline__ void atomicMaxF(float* addr, float v) {
    if (v >= 0.0f) atomicMax((int*)addr, __float_as_int(v));
    else           atomicMin((unsigned int*)addr, __float_as_uint(v));
}

// ---------------- elementwise kernels ----------------
__global__ void fill_f32(float* __restrict__ p, float v, int n) {
    int i = blockIdx.x * blockDim.x + threadIdx.x;
    if (i < n) p[i] = v;
}

__global__ void cvt_bf16(const float* __restrict__ in, unsigned short* __restrict__ out,
                         int n, int do_relu) {
    int i = blockIdx.x * blockDim.x + threadIdx.x;
    if (i < n) {
        float v = in[i];
        if (do_relu) v = fmaxf(v, 0.0f);
        out[i] = f32_to_bf16(v);
    }
}

// ---------------- W pre-pack into WMMA B-fragment layout ----------------
// W is row-major [256 (K)][256 (Nout)] f32.  Packed as bf16 pairs (one dword
// per VGPR slot): tile (kt in 0..7, nt in 0..15), lane in 0..31, j in 0..7.
//   col   = nt*16 + (lane & 15)
//   khalf = lane >> 4
//   k0    = kt*32 + (j>>2)*16 + khalf*8 + 2*(j&3)   (dword = {W[k0][col], W[k0+1][col]})
__global__ void pack_w(const float* __restrict__ W, unsigned int* __restrict__ Wp) {
    int t = blockIdx.x * blockDim.x + threadIdx.x;
    if (t >= 8 * 16 * 32 * 8) return;
    int j    =  t        & 7;
    int lane = (t >> 3)  & 31;
    int nt   = (t >> 8)  & 15;
    int kt   =  t >> 12;
    int col  = nt * 16 + (lane & 15);
    int k0   = kt * 32 + (j >> 2) * 16 + (lane >> 4) * 8 + 2 * (j & 3);
    unsigned int lo = f32_to_bf16(W[(size_t)k0       * FEAT + col]);
    unsigned int hi = f32_to_bf16(W[(size_t)(k0 + 1) * FEAT + col]);
    Wp[t] = lo | (hi << 16);
}

// ---------------- WMMA projection GEMM ----------------
// One wave (32 threads) computes a 16-row x 64-col tile of
//   out = Xbf[rows,256] @ W[256,256] + bias
// grid = (rows/16, 4 col-groups, 2 {src,dst})
__global__ __launch_bounds__(32)
void gemm_proj(const unsigned short* __restrict__ Xbf,
               const unsigned int* __restrict__ WsP, const unsigned int* __restrict__ WdP,
               const float* __restrict__ bs, const float* __restrict__ bd,
               float* __restrict__ fs, float* __restrict__ fd, int rows) {
    const int mTile = blockIdx.x;
    const int ng    = blockIdx.y;            // 4 N-tiles (64 cols)
    const int which = blockIdx.z;            // 0 = src proj, 1 = dst proj
    const unsigned int* __restrict__ Wp   = which ? WdP : WsP;
    const float*        __restrict__ bias = which ? bd  : bs;
    float*              __restrict__ out  = which ? fd  : fs;

    const int lane    = threadIdx.x;
    const int row     = mTile * 16 + (lane & 15);
    const int halfoff = (lane >> 4) * 8;     // A-matrix K half select

    v8f acc[4];
    const v8f vzero = {0.f,0.f,0.f,0.f,0.f,0.f,0.f,0.f};
#pragma unroll
    for (int t = 0; t < 4; ++t) acc[t] = vzero;

    for (int kt = 0; kt < 8; ++kt) {         // K = 256 in 8 steps of 32
        // ---- A fragment: 16x32 bf16, lane-local 8 dwords ----
        union { unsigned int u[8]; v16bf v; } A;
        const unsigned short* xr = Xbf + (size_t)row * FEAT + kt * 32;
#pragma unroll
        for (int j = 0; j < 4; ++j)
            A.u[j]     = *(const unsigned int*)(xr + halfoff + 2 * j);
#pragma unroll
        for (int j = 0; j < 4; ++j)
            A.u[4 + j] = *(const unsigned int*)(xr + 16 + halfoff + 2 * j);

#pragma unroll
        for (int t = 0; t < 4; ++t) {
            int nt = ng * 4 + t;
            union { uint4 q[2]; v16bf v; } Bf;
            const uint4* bp = (const uint4*)(Wp + ((size_t)(kt * 16 + nt) * 32 + lane) * 8);
            Bf.q[0] = bp[0];
            Bf.q[1] = bp[1];
            acc[t] = __builtin_amdgcn_wmma_f32_16x16x32_bf16(
                         false, A.v, false, Bf.v, (short)0, acc[t], false, false);
        }
    }

    // ---- store with fused bias; C/D layout: lane = col, VGPR i = row ----
    const int rbase = mTile * 16 + (lane >> 4) * 8;
#pragma unroll
    for (int t = 0; t < 4; ++t) {
        int col  = ng * 64 + t * 16 + (lane & 15);
        float bv = bias[col];
#pragma unroll
        for (int i = 0; i < 8; ++i) {
            int r = rbase + i;
            if (r < rows) out[(size_t)r * FEAT + col] = acc[t][i] + bv;
        }
    }
}

// ---------------- edge kernels ----------------
// score[e,h] = sum_d attn[h,d] * leaky_relu(fs[src,h,d] + fd[dst,h,d])
__global__ void edge_score(const float* __restrict__ fs, const float* __restrict__ fd,
                           const int* __restrict__ esrc, const int* __restrict__ edst,
                           const float* __restrict__ attn,
                           float* __restrict__ score, float* __restrict__ mx) {
    int t = blockIdx.x * blockDim.x + threadIdx.x;
    if (t >= E_EDGES * NHEAD) return;
    int e = t >> 2, h = t & 3;
    int s = esrc[e], d = edst[e];
    const float4* a = (const float4*)(fs + (size_t)s * FEAT + h * DHEAD);
    const float4* b = (const float4*)(fd + (size_t)d * FEAT + h * DHEAD);
    const float4* w = (const float4*)(attn + h * DHEAD);
    float acc = 0.0f;
#pragma unroll
    for (int j = 0; j < DHEAD / 4; ++j) {
        float4 av = a[j], bv = b[j], wv = w[j];
        float x0 = av.x + bv.x, x1 = av.y + bv.y, x2 = av.z + bv.z, x3 = av.w + bv.w;
        x0 = x0 > 0.f ? x0 : NEG_SLOPE * x0;
        x1 = x1 > 0.f ? x1 : NEG_SLOPE * x1;
        x2 = x2 > 0.f ? x2 : NEG_SLOPE * x2;
        x3 = x3 > 0.f ? x3 : NEG_SLOPE * x3;
        acc += x0 * wv.x + x1 * wv.y + x2 * wv.z + x3 * wv.w;
    }
    score[t] = acc;
    atomicMaxF(mx + (size_t)d * NHEAD + h, acc);
}

// ex = exp(score - mx[dst]); denom[dst] += ex   (score buffer is overwritten with ex)
__global__ void edge_exp(const int* __restrict__ edst, float* __restrict__ score,
                         const float* __restrict__ mx, float* __restrict__ denom) {
    int t = blockIdx.x * blockDim.x + threadIdx.x;
    if (t >= E_EDGES * NHEAD) return;
    int e = t >> 2, h = t & 3;
    int d = edst[e];
    float ex = expf(score[t] - mx[(size_t)d * NHEAD + h]);
    score[t] = ex;
    atomicAdd(denom + (size_t)d * NHEAD + h, ex);
}

// out[dst, f] += alpha[e, f/64] * fs[src, f]   (one 256-thread block per edge)
__global__ __launch_bounds__(256)
void edge_aggregate(const float* __restrict__ fs,
                    const int* __restrict__ esrc, const int* __restrict__ edst,
                    const float* __restrict__ ex, const float* __restrict__ denom,
                    float* __restrict__ out) {
    int e = blockIdx.x;
    int f = threadIdx.x;
    int h = f >> 6;
    int s = esrc[e], d = edst[e];
    float al = ex[(size_t)e * NHEAD + h] /
               fmaxf(denom[(size_t)d * NHEAD + h], 1e-16f);
    atomicAdd(out + (size_t)d * FEAT + f, al * fs[(size_t)s * FEAT + f]);
}

// ---------------- pooling + classifier ----------------
// hm[n,d] = mean_h relu(h2[n,h,d]);  pooled[g,d] = max over nodes of graph g
__global__ void mean_pool(const float* __restrict__ h2, const int* __restrict__ gid,
                          float* __restrict__ pooled) {
    int t = blockIdx.x * blockDim.x + threadIdx.x;
    if (t >= N_NODES * DHEAD) return;
    int n = t >> 6, d = t & 63;
    float s = 0.0f;
#pragma unroll
    for (int h = 0; h < NHEAD; ++h)
        s += fmaxf(h2[(size_t)n * FEAT + h * DHEAD + d], 0.0f);
    atomicMaxF(pooled + (size_t)gid[n] * DHEAD + d, 0.25f * s);
}

__global__ void classify(const float* __restrict__ pooled, const float* __restrict__ Wc,
                         const float* __restrict__ bc, float* __restrict__ out) {
    int t = threadIdx.x;
    if (t >= B_GRAPHS * NCLS) return;
    int b = t / NCLS, c = t % NCLS;
    float acc = bc[c];
#pragma unroll 8
    for (int d = 0; d < DHEAD; ++d)
        acc += pooled[(size_t)b * DHEAD + d] * Wc[(size_t)d * NCLS + c];
    out[t] = acc;
}

// ---------------- launcher ----------------
extern "C" void kernel_launch(void* const* d_in, const int* in_sizes, int n_in,
                              void* d_out, int out_size, void* d_ws, size_t ws_size,
                              hipStream_t stream) {
    const float* h     = (const float*)d_in[0];
    const int*   esrc  = (const int*)  d_in[1];
    const int*   edst  = (const int*)  d_in[2];
    const int*   gid   = (const int*)  d_in[3];
    const float* W1s   = (const float*)d_in[4];
    const float* b1s   = (const float*)d_in[5];
    const float* W1d   = (const float*)d_in[6];
    const float* b1d   = (const float*)d_in[7];
    const float* attn1 = (const float*)d_in[8];
    const float* W2s   = (const float*)d_in[9];
    const float* b2s   = (const float*)d_in[10];
    const float* W2d   = (const float*)d_in[11];
    const float* b2d   = (const float*)d_in[12];
    const float* attn2 = (const float*)d_in[13];
    const float* Wc    = (const float*)d_in[14];
    const float* bc    = (const float*)d_in[15];
    float* out = (float*)d_out;

    // --- carve workspace ---
    char* ws = (char*)d_ws;
    size_t off = 0;
    auto carve = [&](size_t bytes) -> void* {
        void* p = ws + off;
        off += (bytes + 255) & ~(size_t)255;
        return p;
    };
    float*          fs     = (float*)         carve((size_t)N_NODES * FEAT * 4);
    float*          fd     = (float*)         carve((size_t)N_NODES * FEAT * 4);
    float*          hbuf   = (float*)         carve((size_t)N_NODES * FEAT * 4);
    unsigned short* xbf    = (unsigned short*)carve((size_t)N_NODES * FEAT * 2);
    unsigned int*   wsp    = (unsigned int*)  carve((size_t)FEAT * FEAT * 2);
    unsigned int*   wdp    = (unsigned int*)  carve((size_t)FEAT * FEAT * 2);
    float*          score  = (float*)         carve((size_t)E_EDGES * NHEAD * 4);
    float*          mx     = (float*)         carve((size_t)N_NODES * NHEAD * 4);
    float*          denom  = (float*)         carve((size_t)N_NODES * NHEAD * 4);
    float*          pooled = (float*)         carve((size_t)B_GRAPHS * DHEAD * 4);

    const int TPB = 256;
    const int nfeat  = N_NODES * FEAT;            // 12.8M
    const int nheads = N_NODES * NHEAD;           // 200K
    const int nedgeh = E_EDGES * NHEAD;           // 1.6M
    const int npackw = 8 * 16 * 32 * 8;           // 32768 dwords

    // layer-0 input -> bf16
    cvt_bf16<<<(nfeat + TPB - 1) / TPB, TPB, 0, stream>>>(h, xbf, nfeat, 0);

    const float* Ws[2]    = {W1s, W2s};
    const float* Wd[2]    = {W1d, W2d};
    const float* Bs[2]    = {b1s, b2s};
    const float* Bd[2]    = {b1d, b2d};
    const float* attn[2]  = {attn1, attn2};

    for (int layer = 0; layer < 2; ++layer) {
        // pack weights to WMMA B layout
        pack_w<<<(npackw + TPB - 1) / TPB, TPB, 0, stream>>>(Ws[layer], wsp);
        pack_w<<<(npackw + TPB - 1) / TPB, TPB, 0, stream>>>(Wd[layer], wdp);

        // projections via WMMA
        gemm_proj<<<dim3(N_NODES / 16, 4, 2), 32, 0, stream>>>(
            xbf, wsp, wdp, Bs[layer], Bd[layer], fs, fd, N_NODES);

        // softmax state + aggregation target
        fill_f32<<<(nheads + TPB - 1) / TPB, TPB, 0, stream>>>(mx, -INFINITY, nheads);
        fill_f32<<<(nheads + TPB - 1) / TPB, TPB, 0, stream>>>(denom, 0.0f, nheads);
        fill_f32<<<(nfeat + TPB - 1) / TPB, TPB, 0, stream>>>(hbuf, 0.0f, nfeat);

        // edge phase
        edge_score<<<(nedgeh + TPB - 1) / TPB, TPB, 0, stream>>>(
            fs, fd, esrc, edst, attn[layer], score, mx);
        edge_exp<<<(nedgeh + TPB - 1) / TPB, TPB, 0, stream>>>(edst, score, mx, denom);
        edge_aggregate<<<E_EDGES, 256, 0, stream>>>(fs, esrc, edst, score, denom, hbuf);

        if (layer == 0) {
            // relu + bf16 convert: layer-1 output feeds layer 2
            cvt_bf16<<<(nfeat + TPB - 1) / TPB, TPB, 0, stream>>>(hbuf, xbf, nfeat, 1);
        }
    }

    // graph pooling + classifier
    fill_f32<<<(B_GRAPHS * DHEAD + TPB - 1) / TPB, TPB, 0, stream>>>(
        pooled, -INFINITY, B_GRAPHS * DHEAD);
    mean_pool<<<(N_NODES * DHEAD + TPB - 1) / TPB, TPB, 0, stream>>>(hbuf, gid, pooled);
    classify<<<1, 256, 0, stream>>>(pooled, Wc, bc, out);
}
